// ClusteringModule_26551487824507
// MI455X (gfx1250) — matrix operand order repaired and checked by reading, not compiled, not run on hardware
//
#include <hip/hip_runtime.h>
#include <stdint.h>

// Problem constants (from reference)
#define M_TOT   2048      // BN*SN = 32*64
#define K_TOT   65536     // IN_DIM
#define N_TOT   256       // NDF
#define NCLUST  100

// GEMM tiling
#define BM      64
#define BNT     64
#define BK      64        // f32 K elements staged per LDS buffer
#define NSPLIT  4         // split-K factor (deterministic, partials in d_ws)
#define KS      (K_TOT / NSPLIT)   // 16384 K per split
#define LDA     72        // LDS row stride in bf16 elems (64 + 8 pad; 144B, 16B aligned)
#define THREADS 128       // 4 waves (wave32)

typedef __attribute__((ext_vector_type(16))) __bf16 v16bf;
typedef __attribute__((ext_vector_type(8)))  float  v8f;

// Pack two f32 -> {bf16(lo), bf16(hi)}.
// Preferred: single v_cvt_pk_bf16_f32 (RNE). Fallback: +0x8000 bias + v_perm_b32.
static __device__ __forceinline__ unsigned int pk_bf16(float lo, float hi) {
#if __has_builtin(__builtin_amdgcn_cvt_pk_bf16_f32)
    typedef __attribute__((ext_vector_type(2))) __bf16 v2bf;
    union { v2bf v; unsigned int u; } cv;
    cv.v = __builtin_amdgcn_cvt_pk_bf16_f32(lo, hi);   // dst.lo = cvt(lo), dst.hi = cvt(hi)
    return cv.u;
#else
    return __builtin_amdgcn_perm(__float_as_uint(hi) + 0x8000u,
                                 __float_as_uint(lo) + 0x8000u,
                                 0x07060302u);
#endif
}

__global__ __launch_bounds__(THREADS)
void emb_gemm_wmma(const float* __restrict__ A,     // z_roi  [2048, 65536]
                   const float* __restrict__ B,     // w_emb  [65536, 256]
                   float* __restrict__ PT)          // partials [NSPLIT, 2048, 256]
{
    __shared__ unsigned short lA[2][BM  * LDA];     // A tile, row-major (m, k), bf16
    __shared__ unsigned short lB[2][BNT * LDA];     // B tile, TRANSPOSED (n, k), bf16

    const int tid   = threadIdx.x;
    const int m0    = blockIdx.x * BM;
    const int n0    = blockIdx.y * BNT;
    const int kbase = blockIdx.z * KS;

    const int wave = tid >> 5;
    const int lane = tid & 31;
    const int h    = lane >> 4;      // half-wave select
    const int ml   = lane & 15;
    const int wm   = wave >> 1;      // 0..1 : wave row in 2x2 wave grid
    const int wn   = wave & 1;       // 0..1 : wave col

    // A staging map: 16 float4 per 64-wide row, 8 rows base
    const int c4   = tid & 15;
    const int rb   = tid >> 4;       // 0..7
    // B staging map: lane -> column pair (coalesced b64 across n), 4 k-groups of 16
    const int nh   = tid & 31;       // column pair: columns 2*nh, 2*nh+1
    const int kh   = tid >> 5;       // 0..3 : k-group of 16

    const v8f vzero = {0.f,0.f,0.f,0.f,0.f,0.f,0.f,0.f};
    v8f acc[2][2];
    acc[0][0] = vzero; acc[0][1] = vzero; acc[1][0] = vzero; acc[1][1] = vzero;

    float4 aReg[8];
    float2 bReg[16];

    // ---- prologue: fetch k-tile 0 and stage into LDS buffer 0 ----
    #pragma unroll
    for (int i = 0; i < 8; ++i)
        aReg[i] = *(const float4*)&A[(size_t)(m0 + rb + 8*i) * K_TOT + kbase + 4*c4];
    #pragma unroll
    for (int q = 0; q < 16; ++q)
        bReg[q] = *(const float2*)&B[(size_t)(kbase + kh*16 + q) * N_TOT + n0 + 2*nh];

    #pragma unroll
    for (int i = 0; i < 8; ++i) {
        const float4 a4 = aReg[i];
        uint2 pk;
        pk.x = pk_bf16(a4.x, a4.y);
        pk.y = pk_bf16(a4.z, a4.w);
        *(uint2*)&lA[0][(rb + 8*i)*LDA + 4*c4] = pk;
    }
    #pragma unroll
    for (int p = 0; p < 4; ++p) {
        uint2 w0, w1;   // column 2*nh (x lanes), column 2*nh+1 (y lanes); 4 consecutive k each
        w0.x = pk_bf16(bReg[4*p+0].x, bReg[4*p+1].x);
        w0.y = pk_bf16(bReg[4*p+2].x, bReg[4*p+3].x);
        w1.x = pk_bf16(bReg[4*p+0].y, bReg[4*p+1].y);
        w1.y = pk_bf16(bReg[4*p+2].y, bReg[4*p+3].y);
        *(uint2*)&lB[0][(2*nh+0)*LDA + kh*16 + 4*p] = w0;
        *(uint2*)&lB[0][(2*nh+1)*LDA + kh*16 + 4*p] = w1;
    }
    __syncthreads();

    // ---- main K loop over this split, double-buffered & software-pipelined ----
    for (int kt = 0; kt < KS; kt += BK) {
        const int  cur  = (kt / BK) & 1;
        const int  nxt  = cur ^ 1;
        const bool more = (kt + BK) < KS;

        if (more) {
            const int kg = kbase + kt + BK;
            #pragma unroll
            for (int i = 0; i < 8; ++i)
                aReg[i] = *(const float4*)&A[(size_t)(m0 + rb + 8*i) * K_TOT + kg + 4*c4];
            #pragma unroll
            for (int q = 0; q < 16; ++q)
                bReg[q] = *(const float2*)&B[(size_t)(kg + kh*16 + q) * N_TOT + n0 + 2*nh];
            if (kt + 2*BK < KS) { // warm L2 two stages ahead -> global_prefetch_b8
                __builtin_prefetch(&A[(size_t)(m0 + rb) * K_TOT + kbase + kt + 2*BK + 4*c4], 0, 1);
                __builtin_prefetch(&B[(size_t)(kbase + kt + 2*BK + kh*16) * N_TOT + n0 + 2*nh], 0, 1);
            }
        }

        // compute on current buffer: 2 WMMA K-steps of 32
        const unsigned short* As = lA[cur];
        const unsigned short* Bs = lB[cur];
        #pragma unroll
        for (int k32 = 0; k32 < BK; k32 += 32) {
            v16bf af[2], bfr[2];
            #pragma unroll
            for (int i = 0; i < 2; ++i) {
                // 16-bit A 16x32 layout: lane (h,ml) holds K chunks [8h,8h+8) and [16+8h,16+8h+8)
                const int row = wm*32 + i*16 + ml;
                union { uint4 q[2]; v16bf v; } u;
                u.q[0] = *(const uint4*)&As[row*LDA + k32 +      8*h];
                u.q[1] = *(const uint4*)&As[row*LDA + k32 + 16 + 8*h];
                af[i] = u.v;
            }
            #pragma unroll
            for (int j = 0; j < 2; ++j) {
                // B 32x16 layout: lane (h,ml) holds column ml, K chunks [8h,8h+8) and [16+8h,16+8h+8)
                const int col = wn*32 + j*16 + ml;
                union { uint4 q[2]; v16bf v; } u;
                u.q[0] = *(const uint4*)&Bs[col*LDA + k32 +      8*h];
                u.q[1] = *(const uint4*)&Bs[col*LDA + k32 + 16 + 8*h];
                bfr[j] = u.v;
            }
            #pragma unroll
            for (int i = 0; i < 2; ++i)
                #pragma unroll
                for (int j = 0; j < 2; ++j)
                    acc[i][j] = __builtin_amdgcn_wmma_f32_16x16x32_bf16(
                        false, af[i], false, bfr[j], (short)0, acc[i][j], false, false);
        }

        if (more) {
            #pragma unroll
            for (int i = 0; i < 8; ++i) {
                const float4 a4 = aReg[i];
                uint2 pk;
                pk.x = pk_bf16(a4.x, a4.y);
                pk.y = pk_bf16(a4.z, a4.w);
                *(uint2*)&lA[nxt][(rb + 8*i)*LDA + 4*c4] = pk;
            }
            #pragma unroll
            for (int p = 0; p < 4; ++p) {
                uint2 w0, w1;
                w0.x = pk_bf16(bReg[4*p+0].x, bReg[4*p+1].x);
                w0.y = pk_bf16(bReg[4*p+2].x, bReg[4*p+3].x);
                w1.x = pk_bf16(bReg[4*p+0].y, bReg[4*p+1].y);
                w1.y = pk_bf16(bReg[4*p+2].y, bReg[4*p+3].y);
                *(uint2*)&lB[nxt][(2*nh+0)*LDA + kh*16 + 4*p] = w0;
                *(uint2*)&lB[nxt][(2*nh+1)*LDA + kh*16 + 4*p] = w1;
            }
        }
        __syncthreads();
    }

    // ---- epilogue: store f32 partial tile (bias/mask applied at combine) ----
    // C/D f32 16x16 layout: lane col = ml, VGPR r -> row r + 8h
    float* P = PT + (size_t)blockIdx.z * M_TOT * N_TOT;
    #pragma unroll
    for (int j = 0; j < 2; ++j) {
        const int col = n0 + wn*32 + j*16 + ml;
        #pragma unroll
        for (int i = 0; i < 2; ++i) {
            #pragma unroll
            for (int r = 0; r < 8; ++r) {
                const int row = m0 + wm*32 + i*16 + 8*h + r;
                P[(size_t)row * N_TOT + col] = acc[i][j][r];
            }
        }
    }
}

// Kernel 2: combine split-K partials + bias + mask -> z_all, then student-t
// soft assignment + row-normalize + argmax. One wave per row. ALPHA=1 => s_tmp=(1+dist)^-1.
__global__ __launch_bounds__(128)
void combine_student_t(const float* __restrict__ PT,        // [NSPLIT,2048,256]
                       const float* __restrict__ bias,      // [256]
                       const unsigned char* __restrict__ mask,
                       const float* __restrict__ C,         // centroids [100,256]
                       float* __restrict__ Z,               // z_all [2048,256]
                       float* __restrict__ S,               // [2048,100]
                       float* __restrict__ Cout)            // [2048] (as float)
{
    __shared__ float zrow[4][N_TOT];
    const int tid  = threadIdx.x;
    const int w    = tid >> 5;
    const int lane = tid & 31;
    const int row  = blockIdx.x * 4 + w;

    const bool mrow = mask[row] != 0;
    const size_t MN = (size_t)M_TOT * N_TOT;

    for (int d = lane; d < N_TOT; d += 32) {
        const size_t idx = (size_t)row * N_TOT + d;
        const float z = ((PT[idx] + PT[idx + MN]) + (PT[idx + 2*MN] + PT[idx + 3*MN]))
                        + bias[d];
        zrow[w][d] = z;                       // unmasked z feeds student-t (matches reference)
        Z[idx] = mrow ? z : 0.0f;             // z_all is masked
    }
    __syncthreads();

    float stv[4]; int cid[4];
    float psum = 0.f, best_v = -1.0f; int best_i = 0;
    int t = 0;
    for (int c0 = 0; c0 < NCLUST; c0 += 32, ++t) {
        const int c = c0 + lane;
        float v = 0.f;
        if (c < NCLUST) {
            const float* cp = &C[(size_t)c * N_TOT];
            float d2 = 0.f;
            #pragma unroll 8
            for (int d = 0; d < N_TOT; ++d) {
                const float df = zrow[w][d] - cp[d];
                d2 = fmaf(df, df, d2);
            }
            const float dist = sqrtf(fmaxf(d2, 0.f));
            v = 1.0f / (1.0f + dist);         // (1 + dist/ALPHA)^(-(ALPHA+1)/2), ALPHA=1
            psum += v;
            if (v > best_v) { best_v = v; best_i = c; }  // strict > keeps first (lowest) index
        }
        stv[t] = v;
        cid[t] = (c < NCLUST) ? c : -1;
    }

    // wave32 butterfly reductions: sum + first-index argmax
    #pragma unroll
    for (int off = 16; off > 0; off >>= 1) {
        psum += __shfl_xor(psum, off, 32);
        const float ov = __shfl_xor(best_v, off, 32);
        const int   oi = __shfl_xor(best_i, off, 32);
        if (ov > best_v || (ov == best_v && oi < best_i)) { best_v = ov; best_i = oi; }
    }

    const float inv = (mrow && psum > 0.f) ? (1.0f / psum) : 0.f;
    #pragma unroll
    for (int q = 0; q < 4; ++q)
        if (cid[q] >= 0)
            S[(size_t)row * NCLUST + cid[q]] = stv[q] * inv;
    if (lane == 0)
        Cout[row] = mrow ? (float)best_i : 0.0f;
}

extern "C" void kernel_launch(void* const* d_in, const int* in_sizes, int n_in,
                              void* d_out, int out_size, void* d_ws, size_t ws_size,
                              hipStream_t stream) {
    (void)in_sizes; (void)n_in; (void)out_size; (void)ws_size;
    const float*         z_roi     = (const float*)d_in[0];
    const unsigned char* mask      = (const unsigned char*)d_in[1];
    const float*         w_emb     = (const float*)d_in[2];
    const float*         b_emb     = (const float*)d_in[3];
    const float*         centroids = (const float*)d_in[4];

    float* out   = (float*)d_out;
    float* out_z = out;                                   // [2048,256]
    float* out_s = out_z + (size_t)M_TOT * N_TOT;         // [2048,100]
    float* out_c = out_s + (size_t)M_TOT * NCLUST;        // [2048]

    float* partials = (float*)d_ws;                       // NSPLIT * 2 MB = 8 MB

    dim3 g1(M_TOT / BM, N_TOT / BNT, NSPLIT);             // 32 x 4 x 4 = 512 blocks
    emb_gemm_wmma<<<g1, THREADS, 0, stream>>>(z_roi, w_emb, partials);

    combine_student_t<<<M_TOT / 4, 128, 0, stream>>>(partials, b_emb, mask, centroids,
                                                     out_z, out_s, out_c);
}
